// EmbeddingNet_85658827751855
// MI455X (gfx1250) — matrix operand order repaired and checked by reading, not compiled, not run on hardware
//
#include <hip/hip_runtime.h>
#include <hip/hip_bf16.h>
#include <math.h>

typedef __attribute__((ext_vector_type(16))) _Float16 v16h;
typedef __attribute__((ext_vector_type(8)))  _Float16 v8h;
typedef __attribute__((ext_vector_type(8)))  float    v8f;

#define EPS 1e-5f

// ---------------------------------------------------------------------------
// A-fragment k ordering used by this code (contiguous, same as the B loads):
//   lane <16 : k = {0..15};  lane >=16 : k = {16..31}
__device__ __host__ inline int a_k_of(int lane, int j) {
    int base = (lane >= 16) ? 16 : 0;
    return j + base;
}

// ---------------------------------------------------------------------------
// Pack node_w (4,32,64,3,3) fp32 -> f16 A fragments:
//   dst[((((i*9+r)*2+c)*2+t)*32+lane)*16 + j] = W[i][t*16+(lane&15)][c*32+k][r]
__global__ __launch_bounds__(256) void pack_node_w(const float* __restrict__ w,
                                                   _Float16* __restrict__ dst) {
    int tid = blockIdx.x * 256 + threadIdx.x;
    if (tid >= 4 * 9 * 2 * 2 * 32 * 16) return;
    int j    = tid & 15;
    int lane = (tid >> 4) & 31;
    int rest = tid >> 9;
    int t    = rest & 1;
    int c    = (rest >> 1) & 1;
    int rc   = rest >> 2;
    int r    = rc % 9;
    int i    = rc / 9;
    int co = t * 16 + (lane & 15);
    int ci = c * 32 + a_k_of(lane, j);
    float v = w[(((size_t)(i * 32 + co) * 64 + ci) * 9) + r];
    dst[tid] = (_Float16)v;
}

// Pack final_w (8,32,3,3) -> dst[(r*32+lane)*16 + j], rows 8..15 zero.
__global__ __launch_bounds__(256) void pack_final_w(const float* __restrict__ w,
                                                    _Float16* __restrict__ dst) {
    int tid = blockIdx.x * 256 + threadIdx.x;
    if (tid >= 9 * 32 * 16) return;
    int j    = tid & 15;
    int lane = (tid >> 4) & 31;
    int r    = tid >> 9;
    int co   = lane & 15;
    int ci   = a_k_of(lane, j);
    float v  = (co < 8) ? w[(((size_t)(co * 32 + ci)) * 9) + r] : 0.0f;
    dst[tid] = (_Float16)v;
}

// ---------------------------------------------------------------------------
// NCHW fp32 (32 channels) -> f16 NHWC slot: dst[pixel*64 + coff + c]
__global__ __launch_bounds__(256) void to_nhwc_f16(const float* __restrict__ src,
                                                   _Float16* __restrict__ dst,
                                                   int coff) {
    size_t tid = (size_t)blockIdx.x * 256 + threadIdx.x;   // 16*128*128*32
    if (tid >= (size_t)16 * 128 * 128 * 32) return;
    int c      = (int)(tid & 31);
    size_t p   = tid >> 5;                  // pixel = b*16384 + y*128 + x
    size_t b   = p >> 14;
    size_t rem = p & 16383;
    dst[p * 64 + coff + c] = (_Float16)src[(b * 32 + c) * 16384 + rem];
}

// ---------------------------------------------------------------------------
// Stage 3 halo rows of a 64-channel NHWC f16 tensor into LDS via
// GLOBAL_LOAD_ASYNC_TO_LDS_B128 (GVS mode). lds[(row*130+col)*64 + ci].
__device__ inline void stage_rows_async(_Float16* lds, const _Float16* In,
                                        int b, int y, int tid) {
    unsigned long long base = (unsigned long long)In;
    v8h z = {};
    // 3 rows x 130 cols x 128B = 3120 chunks of 16B
    for (int c = tid; c < 3 * 1040; c += 256) {
        int row = c / 1040;
        int cc  = c % 1040;
        int col = cc >> 3;
        int sub = cc & 7;
        int yy  = y + row - 1;
        int x   = col - 1;
        int eoff = ((row * 130 + col) * 64) + sub * 8;
        if ((unsigned)yy < 128u && (unsigned)x < 128u) {
            unsigned loff = (unsigned)(size_t)&lds[eoff];   // low 32b = LDS addr
            unsigned goff = (unsigned)((((b * 128 + yy) * 128 + x) * 64 + sub * 8) * 2);
            asm volatile("global_load_async_to_lds_b128 %0, %1, %2"
                         :: "v"(loff), "v"(goff), "s"(base) : "memory");
        } else {
            *(v8h*)&lds[eoff] = z;                          // zero halo/pad
        }
    }
    asm volatile("s_wait_asynccnt 0" ::: "memory");
    __syncthreads();
}

// ---------------------------------------------------------------------------
// Node layer: y = relu(bn(conv3x3(In64, W)))   (64 -> 32 channels)
// In: NHWC f16 [16][128][128][64]; Out: NHWC f16, writes channels 0..31.
__global__ __launch_bounds__(256) void node_conv_kernel(
    const _Float16* __restrict__ In,
    const _Float16* __restrict__ Wp,   // packed [9][2][2][32][16]
    const float* __restrict__ gamma, const float* __restrict__ beta,
    const float* __restrict__ mean,  const float* __restrict__ var,
    _Float16* __restrict__ Out)
{
    __shared__ _Float16 lds[3 * 130 * 64] __attribute__((aligned(32)));
    const int wg  = blockIdx.x;
    const int b   = wg >> 7;
    const int y   = wg & 127;
    const int tid = threadIdx.x;

    stage_rows_async(lds, In, b, y, tid);

    const int lane  = tid & 31;
    const int wave  = tid >> 5;     // 0..7 -> 16-pixel column each
    const int xbase = wave * 16;
    const int n     = lane & 15;
    const int khi   = lane >> 4;    // 0/1: B half (K 0-15 vs 16-31)

    v8f acc0 = {};
    v8f acc1 = {};

#pragma unroll
    for (int r = 0; r < 9; ++r) {
        const int dy = r / 3, dx = r % 3;
#pragma unroll
        for (int c = 0; c < 2; ++c) {
            const int col = xbase + n + dx;                 // 0..129
            const int cib = c * 32 + khi * 16;              // 16-aligned
            v16h bf = *(const v16h*)&lds[((size_t)(dy * 130 + col)) * 64 + cib];
            v16h a0 = *(const v16h*)(Wp + (((size_t)((r * 2 + c) * 2 + 0) * 32 + lane) * 16));
            v16h a1 = *(const v16h*)(Wp + (((size_t)((r * 2 + c) * 2 + 1) * 32 + lane) * 16));
            acc0 = __builtin_amdgcn_wmma_f32_16x16x32_f16(false, a0, false, bf,
                                                          (short)0, acc0, false, false);
            acc1 = __builtin_amdgcn_wmma_f32_16x16x32_f16(false, a1, false, bf,
                                                          (short)0, acc1, false, false);
        }
    }

    // D tile: VGPR v, lane -> (co = t*16 + 8*khi + v, col n). Fused BN+ReLU,
    // packed f16 NHWC store: one b128 per tile per lane.
    const int x = xbase + n;
    _Float16* outp = Out + (((size_t)(b * 128 + y) * 128 + x) * 64);
#pragma unroll
    for (int t = 0; t < 2; ++t) {
        v8f acc = t ? acc1 : acc0;
        v8h o;
#pragma unroll
        for (int v = 0; v < 8; ++v) {
            int co  = t * 16 + 8 * khi + v;
            float s = gamma[co] * rsqrtf(var[co] + EPS);
            float f = fmaxf(acc[v] * s + (beta[co] - mean[co] * s), 0.0f);
            o[v] = (_Float16)f;
        }
        *(v8h*)(outp + t * 16 + 8 * khi) = o;
    }
}

// ---------------------------------------------------------------------------
// Final layer: y = tanh(bn(conv3x3(In[ch 0..31], Wf)))  -> fp32 NCHW 8ch
__global__ __launch_bounds__(256) void final_conv_kernel(
    const _Float16* __restrict__ In,   // NHWC f16, uses channels 0..31
    const _Float16* __restrict__ Wp,   // packed [9][32][16]
    const float* __restrict__ gamma, const float* __restrict__ beta,
    const float* __restrict__ mean,  const float* __restrict__ var,
    float* __restrict__ Y)             // [16][8][128][128]
{
    __shared__ _Float16 lds[3 * 130 * 64] __attribute__((aligned(32)));
    const int wg  = blockIdx.x;
    const int b   = wg >> 7;
    const int y   = wg & 127;
    const int tid = threadIdx.x;

    stage_rows_async(lds, In, b, y, tid);

    const int lane  = tid & 31;
    const int wave  = tid >> 5;
    const int xbase = wave * 16;
    const int n     = lane & 15;
    const int khi   = lane >> 4;

    v8f acc = {};
#pragma unroll
    for (int r = 0; r < 9; ++r) {
        const int dy = r / 3, dx = r % 3;
        const int col = xbase + n + dx;
        v16h bf = *(const v16h*)&lds[((size_t)(dy * 130 + col)) * 64 + khi * 16];
        v16h af = *(const v16h*)(Wp + (((size_t)r * 32 + lane) * 16));
        acc = __builtin_amdgcn_wmma_f32_16x16x32_f16(false, af, false, bf,
                                                     (short)0, acc, false, false);
    }

    const int x = xbase + n;
#pragma unroll
    for (int v = 0; v < 8; ++v) {
        int co = v + 8 * khi;
        if (co < 8) {
            float s = gamma[co] * rsqrtf(var[co] + EPS);
            float o = acc[v] * s + (beta[co] - mean[co] * s);
            Y[((size_t)b * 8 + co) * 16384 + (size_t)y * 128 + x] = tanhf(o);
        }
    }
}

// ---------------------------------------------------------------------------
// Depthwise ConvTranspose2d(k=4, s=2, p=1) + per-row/col ramps, fused.
__global__ __launch_bounds__(256) void upsample_kernel(
    const float* __restrict__ Yin,   // [16][8][128][128]
    const float* __restrict__ up_w,  // [8][1][4][4]
    float* __restrict__ out)         // [16][8][256][256]
{
    size_t idx = (size_t)blockIdx.x * 256 + threadIdx.x;
    if (idx >= (size_t)16 * 8 * 256 * 256) return;
    int ox = idx & 255;
    int oy = (int)((idx >> 8) & 255);
    int ch = (int)((idx >> 16) & 7);
    int b  = (int)(idx >> 19);

    const float* w  = up_w + ch * 16;
    const float* in = Yin + ((size_t)b * 8 + ch) * 16384;
    float acc = 0.0f;
#pragma unroll
    for (int ky = 0; ky < 4; ++ky) {
        int ty = oy + 1 - ky;
        if (ty & 1) continue;
        int iy = ty >> 1;
        if ((unsigned)iy >= 128u) continue;
#pragma unroll
        for (int kx = 0; kx < 4; ++kx) {
            int tx = ox + 1 - kx;
            if (tx & 1) continue;
            int ix = tx >> 1;
            if ((unsigned)ix >= 128u) continue;
            acc += in[iy * 128 + ix] * w[ky * 4 + kx];
        }
    }
    if (ch == 0) acc += (float)oy * (1.0f / 256.0f);
    if (ch == 1) acc += (float)ox * (1.0f / 256.0f);
    out[idx] = acc;
}

// ---------------------------------------------------------------------------
extern "C" void kernel_launch(void* const* d_in, const int* in_sizes, int n_in,
                              void* d_out, int out_size, void* d_ws, size_t ws_size,
                              hipStream_t stream) {
    const float* layers      = (const float*)d_in[0];   // [5][16][32][128][128]
    const float* node_w      = (const float*)d_in[1];   // [4][32][64][3][3]
    const float* node_gamma  = (const float*)d_in[2];   // [4][32]
    const float* node_beta   = (const float*)d_in[3];
    const float* node_mean   = (const float*)d_in[4];
    const float* node_var    = (const float*)d_in[5];
    const float* final_w     = (const float*)d_in[6];   // [8][32][3][3]
    const float* final_gamma = (const float*)d_in[7];
    const float* final_beta  = (const float*)d_in[8];
    const float* final_mean  = (const float*)d_in[9];
    const float* final_var   = (const float*)d_in[10];
    const float* up_w        = (const float*)d_in[11];  // [8][1][4][4]

    const size_t BUF = (size_t)16 * 128 * 128 * 64 * 2;  // 33.55 MB f16 NHWC
    char* ws = (char*)d_ws;
    _Float16* wnode = (_Float16*)ws;                     // 73728 f16
    _Float16* wfin  = (_Float16*)(ws + 192 * 1024);      // 4608 f16
    _Float16* buf0  = (_Float16*)(ws + 256 * 1024);
    _Float16* buf1  = (_Float16*)(ws + 256 * 1024 + BUF);
    float*    Yf    = (float*)   (ws + 256 * 1024 + 2 * BUF);  // 8.39 MB
    if (ws_size < 256 * 1024 + 2 * BUF + (size_t)16 * 8 * 128 * 128 * 4) return;

    pack_node_w<<<(73728 + 255) / 256, 256, 0, stream>>>(node_w, wnode);
    pack_final_w<<<(4608 + 255) / 256, 256, 0, stream>>>(final_w, wfin);

    const size_t LPLANE = (size_t)16 * 32 * 128 * 128;
    const int CVT_BLOCKS = (int)(((size_t)16 * 128 * 128 * 32 + 255) / 256);

    _Float16* bufs[2] = {buf0, buf1};
    // layers[0] -> buf0 lower, layers[1] -> buf0 upper
    to_nhwc_f16<<<CVT_BLOCKS, 256, 0, stream>>>(layers, buf0, 0);
    to_nhwc_f16<<<CVT_BLOCKS, 256, 0, stream>>>(layers + LPLANE, buf0, 32);

    for (int i = 0; i < 4; ++i) {
        _Float16* cur = bufs[i & 1];
        _Float16* nxt = bufs[(i + 1) & 1];
        if (i < 3)  // pre-stage layers[i+2] into upper half of next buffer
            to_nhwc_f16<<<CVT_BLOCKS, 256, 0, stream>>>(
                layers + (size_t)(i + 2) * LPLANE, nxt, 32);
        node_conv_kernel<<<2048, 256, 0, stream>>>(
            cur, wnode + (size_t)i * 9 * 2 * 2 * 32 * 16,
            node_gamma + i * 32, node_beta + i * 32,
            node_mean + i * 32, node_var + i * 32, nxt);
    }
    // node 3 wrote buf0 lower half
    final_conv_kernel<<<2048, 256, 0, stream>>>(
        buf0, wfin, final_gamma, final_beta, final_mean, final_var, Yf);

    size_t tot = (size_t)16 * 8 * 256 * 256;
    upsample_kernel<<<(tot + 255) / 256, 256, 0, stream>>>(Yf, up_w, (float*)d_out);
}
